// AgentAttention_1_26276609917671
// MI455X (gfx1250) — matrix operand order repaired and checked by reading, not compiled, not run on hardware
//
#include <hip/hip_runtime.h>
#include <hip/hip_bf16.h>

// ---------------- types ----------------
typedef _Float16 h8   __attribute__((ext_vector_type(8)));
typedef _Float16 v16h __attribute__((ext_vector_type(16)));
typedef float    v8f  __attribute__((ext_vector_type(8)));
typedef float    f4v  __attribute__((ext_vector_type(4)));

#define DEV __device__ __forceinline__

static constexpr int BB = 512, NN = 196, CC = 192, HEADS = 6, AGENT = 49, HD = 32, HW = 14;
static constexpr float SCALE = 0.17677669529663687f; // 32^-0.5

DEV v16h v16h_zero() {
  v16h z;
#pragma unroll
  for (int i = 0; i < 16; ++i) z[i] = (_Float16)0;
  return z;
}
DEV v8f v8f_zero() {
  v8f z;
#pragma unroll
  for (int i = 0; i < 8; ++i) z[i] = 0.f;
  return z;
}

// A fragment (16x32 f16, row-major source [M][K], ld in elements).
// ISA layout: lane m = lane&15, half = lane>>4; elems 0..7 = K[8h..8h+7], 8..15 = K[16+8h..].
DEV v16h frag_a_f16(const _Float16* base, int ld, int m0, int k0, int lane) {
  int mr = lane & 15, hf = lane >> 4;
  const _Float16* p = base + (size_t)(m0 + mr) * ld + k0 + hf * 8;
  h8 c0 = *(const h8*)p;
  h8 c1 = *(const h8*)(p + 16);
  v16h r;
#pragma unroll
  for (int i = 0; i < 8; ++i) { r[i] = c0[i]; r[i + 8] = c1[i]; }
  return r;
}
DEV v16h frag_a_f16_g(const _Float16* base, int ld, int m0, int k0, int lane, int mvalid) {
  int mr = lane & 15;
  if (m0 + mr >= mvalid) return v16h_zero();
  return frag_a_f16(base, ld, m0, k0, lane);
}
// A fragment from f32 source, converted in flight.
DEV v16h frag_a_f32(const float* base, int ld, int m0, int k0, int lane) {
  int mr = lane & 15, hf = lane >> 4;
  const float* p = base + (size_t)(m0 + mr) * ld + k0 + hf * 8;
  f4v c0 = *(const f4v*)p;
  f4v c1 = *(const f4v*)(p + 4);
  f4v c2 = *(const f4v*)(p + 16);
  f4v c3 = *(const f4v*)(p + 20);
  v16h r;
#pragma unroll
  for (int i = 0; i < 4; ++i) {
    r[i] = (_Float16)c0[i]; r[4 + i] = (_Float16)c1[i];
    r[8 + i] = (_Float16)c2[i]; r[12 + i] = (_Float16)c3[i];
  }
  return r;
}
// B fragment (32x16 f16). Source is Bt[N][K] row-major (ld in elements).
// ISA layout: lane n = lane&15, khalf = lane>>4; elems 0..7 = K[8kh..8kh+7], 8..15 = K[16+8kh..].
DEV v16h frag_b_f16(const _Float16* bt, int ld, int n0, int k0, int lane) {
  int nc = lane & 15, kh = lane >> 4;
  const _Float16* p = bt + (size_t)(n0 + nc) * ld + k0 + kh * 8;
  h8 c0 = *(const h8*)p;
  h8 c1 = *(const h8*)(p + 16);
  v16h r;
#pragma unroll
  for (int i = 0; i < 8; ++i) { r[i] = c0[i]; r[i + 8] = c1[i]; }
  return r;
}
DEV v16h frag_b_f16_g(const _Float16* bt, int ld, int n0, int k0, int lane, int nvalid) {
  int nc = lane & 15;
  if (n0 + nc >= nvalid) return v16h_zero();
  return frag_b_f16(bt, ld, n0, k0, lane);
}
DEV v8f wmma16(v16h a, v16h b, v8f c) {
  return __builtin_amdgcn_wmma_f32_16x16x32_f16(false, a, false, b, (short)0, c, false, false);
}

// ---------------- K0: bias precompute (bilinear resize + separable biases) ----------------
DEV void interp_w(int t, int& i0, int& i1, float& w0, float& w1) {
  float pos = 0.5f * (float)t - 0.25f;
  int f0 = (int)floorf(pos);
  float f = pos - (float)f0;
  int a0 = f0, a1 = f0 + 1;
  w0 = 1.f - f; w1 = f;
  if (a0 < 0) { w0 = 0.f; a0 = 0; }
  if (a1 > 6) { w1 = 0.f; a1 = 6; }
  float s = w0 + w1;
  w0 /= s; w1 /= s;
  i0 = a0; i1 = a1;
}

__global__ void bias_kernel(const float* __restrict__ an, const float* __restrict__ na,
                            const float* __restrict__ ah, const float* __restrict__ aw,
                            const float* __restrict__ ha, const float* __restrict__ wa,
                            float* __restrict__ PB, float* __restrict__ AB) {
  int idx = blockIdx.x * 256 + threadIdx.x;
  if (idx >= HEADS * AGENT * NN) return;
  int n = idx % NN;
  int t = idx / NN;
  int a = t % AGENT;
  int h = t / AGENT;
  int y = n / HW, x = n % HW;
  int iy0, iy1, ix0, ix1;
  float wy0, wy1, wx0, wx1;
  interp_w(y, iy0, iy1, wy0, wy1);
  interp_w(x, ix0, ix1, wx0, wx1);
  const float* s1 = an + (size_t)(h * AGENT + a) * 49;
  float r1 = wy0 * (wx0 * s1[iy0 * 7 + ix0] + wx1 * s1[iy0 * 7 + ix1]) +
             wy1 * (wx0 * s1[iy1 * 7 + ix0] + wx1 * s1[iy1 * 7 + ix1]);
  PB[idx] = r1 + ah[(h * AGENT + a) * HW + y] + aw[(h * AGENT + a) * HW + x];
  const float* s2 = na + (size_t)(h * AGENT + a) * 49;
  float r2 = wy0 * (wx0 * s2[iy0 * 7 + ix0] + wx1 * s2[iy0 * 7 + ix1]) +
             wy1 * (wx0 * s2[iy1 * 7 + ix0] + wx1 * s2[iy1 * 7 + ix1]);
  AB[((size_t)h * NN + n) * AGENT + a] = r2 + ha[(h * HW + y) * AGENT + a] + wa[(h * HW + x) * AGENT + a];
}

// ---------------- K0b: weight f32 -> f16 ----------------
__global__ void cvt_kernel(const float* __restrict__ qkvw, const float* __restrict__ projw,
                           _Float16* __restrict__ w16, _Float16* __restrict__ p16) {
  int idx = blockIdx.x * 256 + threadIdx.x;
  if (idx < 576 * 192) w16[idx] = (_Float16)qkvw[idx];
  else {
    int j = idx - 576 * 192;
    if (j < 192 * 192) p16[j] = (_Float16)projw[j];
  }
}

// ---------------- K1: qkv = x @ W^T + b, stored as f16 q/k/v ----------------
__global__ __launch_bounds__(256) void qkv_kernel(const float* __restrict__ x,
                                                  const _Float16* __restrict__ w16,
                                                  const float* __restrict__ bias,
                                                  _Float16* __restrict__ q,
                                                  _Float16* __restrict__ k,
                                                  _Float16* __restrict__ v) {
  int lane = threadIdx.x & 31;
  int job = blockIdx.x * 8 + (threadIdx.x >> 5); // 56448 wave-jobs
  int jm = job / 9, jn = job % 9;
  int m0 = jm * 16;
  v8f acc[4];
#pragma unroll
  for (int t = 0; t < 4; ++t) acc[t] = v8f_zero();
#pragma unroll
  for (int kt = 0; kt < 6; ++kt) {
    int k0 = kt * 32;
    v16h a = frag_a_f32(x, CC, m0, k0, lane);
#pragma unroll
    for (int t = 0; t < 4; ++t) {
      v16h bf = frag_b_f16(w16, CC, jn * 64 + t * 16, k0, lane);
      acc[t] = wmma16(a, bf, acc[t]);
    }
  }
  int hf = lane >> 4, nc = lane & 15;
#pragma unroll
  for (int t = 0; t < 4; ++t) {
    int col = jn * 64 + t * 16 + nc;
    float bv = bias[col];
    int which = col / CC, jj = col % CC;
    _Float16* dst = which == 0 ? q : (which == 1 ? k : v);
#pragma unroll
    for (int r = 0; r < 8; ++r) {
      int m = m0 + hf * 8 + r;
      dst[(size_t)m * CC + jj] = (_Float16)(acc[t][r] + bv);
    }
  }
}

// ---------------- K2: agent pooling (2x2 mean over q) ----------------
__global__ void pool_kernel(const _Float16* __restrict__ qf, _Float16* __restrict__ agent) {
  int idx = blockIdx.x * 256 + threadIdx.x; // B*49*192 exact
  int c = idx % CC;
  int t = idx / CC;
  int a = t % AGENT;
  int b = t / AGENT;
  int ay = a / 7, ax = a % 7;
  float s = 0.f;
#pragma unroll
  for (int dy = 0; dy < 2; ++dy)
#pragma unroll
    for (int dx = 0; dx < 2; ++dx) {
      int n = (2 * ay + dy) * HW + (2 * ax + dx);
      s += (float)qf[((size_t)b * NN + n) * CC + c];
    }
  agent[idx] = (_Float16)(0.25f * s);
}

// ---------------- K3: stage-1 attention per (b,h) ----------------
// S:  float[64][224], A2: f16[64][224], vT: f16[32][224]   (100,352 B dyn LDS)
__global__ __launch_bounds__(128) void stage1_kernel(const _Float16* __restrict__ kf,
                                                     const _Float16* __restrict__ vf,
                                                     const _Float16* __restrict__ agent,
                                                     const float* __restrict__ PB,
                                                     _Float16* __restrict__ avT) {
  int b = blockIdx.x / HEADS, h = blockIdx.x % HEADS;
  extern __shared__ char smem[];
  float* S = (float*)smem;                                        // 64*224 f32
  _Float16* A2 = (_Float16*)(smem + 64 * 224 * 4);                // 64*224 f16
  _Float16* vT = (_Float16*)(smem + 64 * 224 * 4 + 64 * 224 * 2); // 32*224 f16
  int tid = threadIdx.x, lane = tid & 31, wid = tid >> 5;
  const _Float16* kbase = kf + (size_t)b * NN * CC + h * HD;
  const _Float16* vbase = vf + (size_t)b * NN * CC + h * HD;
  const _Float16* abase = agent + (size_t)b * AGENT * CC + h * HD;

  for (int i = tid; i < 32 * 224; i += 128) {
    int d = i / 224, n = i % 224;
    vT[d * 224 + n] = (n < NN) ? vbase[(size_t)n * CC + d] : (_Float16)0;
  }
  // logits = scale * ag @ k^T + PB ; 4 Mtiles x 14 Ntiles, K=32
  for (int j = wid; j < 4 * 14; j += 4) {
    int mt = j / 14, nt = j % 14;
    int m0 = mt * 16, n0 = nt * 16;
    v16h a = frag_a_f16_g(abase, CC, m0, 0, lane, AGENT);
    v16h bf = frag_b_f16_g(kbase, CC, n0, 0, lane, NN);
    v8f c = v8f_zero();
    c = wmma16(a, bf, c);
    int hf = lane >> 4, nc = lane & 15;
#pragma unroll
    for (int r = 0; r < 8; ++r) {
      int m = m0 + hf * 8 + r, n = n0 + nc;
      float val = (m < AGENT && n < NN) ? SCALE * c[r] + PB[(size_t)(h * AGENT + m) * NN + n] : -1e30f;
      S[m * 224 + n] = val;
    }
  }
  __syncthreads();
  if (tid < 64) {
    int m = tid;
    float mx = -3e30f;
    for (int n = 0; n < 224; ++n) mx = fmaxf(mx, S[m * 224 + n]);
    float sum = 0.f;
    for (int n = 0; n < 224; ++n) {
      float e = __expf(S[m * 224 + n] - mx);
      S[m * 224 + n] = e;
      sum += e;
    }
    float inv = 1.f / sum;
    for (int n = 0; n < 224; ++n) A2[m * 224 + n] = (_Float16)(S[m * 224 + n] * inv);
  }
  __syncthreads();
  // agent_v = attn @ v ; store transposed (avT[d][a]) zero-padded in a
  _Float16* avbase = avT + ((size_t)b * HEADS + h) * 32 * 64;
  for (int j = wid; j < 8; j += 4) {
    int mt = j >> 1, nt = j & 1;
    int m0 = mt * 16, n0 = nt * 16;
    v8f c = v8f_zero();
#pragma unroll
    for (int kt = 0; kt < 7; ++kt) {
      v16h a = frag_a_f16(A2, 224, m0, kt * 32, lane);
      v16h bf = frag_b_f16(vT, 224, n0, kt * 32, lane);
      c = wmma16(a, bf, c);
    }
    int hf = lane >> 4, nc = lane & 15;
#pragma unroll
    for (int r = 0; r < 8; ++r) {
      int aI = m0 + hf * 8 + r, d = n0 + nc;
      float val = (aI < AGENT) ? c[r] : 0.f;
      avbase[(size_t)d * 64 + aI] = (_Float16)val;
    }
  }
}

// ---------------- K4: stage-2 + dwconv + projection, fused per batch ----------------
// S2: f32[208][64]; outp: f16[208][192]; A2: f16[208][64]; qh: f16[208][32];
// agb: f16[64][32]; avb: f16[32][64]  => 181,248 B dyn LDS
__global__ __launch_bounds__(256) void stage2_kernel(const _Float16* __restrict__ qf,
                                                     const _Float16* __restrict__ vf,
                                                     const _Float16* __restrict__ agent,
                                                     const _Float16* __restrict__ avT,
                                                     const float* __restrict__ AB,
                                                     const _Float16* __restrict__ pw16,
                                                     const float* __restrict__ proj_b,
                                                     const float* __restrict__ dwc_w,
                                                     const float* __restrict__ dwc_b,
                                                     float* __restrict__ out) {
  int b = blockIdx.x;
  extern __shared__ char smem[];
  float* S2 = (float*)smem;                                // 53,248 B
  _Float16* outp = (_Float16*)(smem + 53248);              // 79,872 B
  _Float16* A2 = (_Float16*)(smem + 133120);               // 26,624 B
  _Float16* qh = (_Float16*)(smem + 159744);               // 13,312 B
  _Float16* agb = (_Float16*)(smem + 173056);              //  4,096 B
  _Float16* avb = (_Float16*)(smem + 177152);              //  4,096 B
  int tid = threadIdx.x, lane = tid & 31, wid = tid >> 5;

  // depthwise 3x3 conv on v + dwc_b -> outp (rows >=196 zero)
  for (int i = tid; i < 208 * 192; i += 256) {
    int n = i / CC, c = i % CC;
    float acc = 0.f;
    if (n < NN) {
      int y = n / HW, x = n % HW;
      acc = dwc_b[c];
#pragma unroll
      for (int dy = 0; dy < 3; ++dy)
#pragma unroll
        for (int dx = 0; dx < 3; ++dx) {
          int yy = y + dy - 1, xc = x + dx - 1;
          if (yy >= 0 && yy < HW && xc >= 0 && xc < HW)
            acc += (float)vf[((size_t)b * NN + yy * HW + xc) * CC + c] * dwc_w[c * 9 + dy * 3 + dx];
        }
    }
    outp[i] = (_Float16)acc;
  }
  __syncthreads();

  for (int h = 0; h < HEADS; ++h) {
    for (int i = tid; i < 208 * 32; i += 256) {
      int n = i >> 5, d = i & 31;
      qh[i] = (n < NN) ? qf[((size_t)b * NN + n) * CC + h * HD + d] : (_Float16)0;
    }
    for (int i = tid; i < 64 * 32; i += 256) {
      int a = i >> 5, d = i & 31;
      agb[i] = (a < AGENT) ? agent[((size_t)b * AGENT + a) * CC + h * HD + d] : (_Float16)0;
    }
    for (int i = tid; i < 32 * 64; i += 256)
      avb[i] = avT[((size_t)b * HEADS + h) * 32 * 64 + i];
    __syncthreads();

    // logits2 = scale * q @ ag^T + AB ; 13 Mtiles x 4 Ntiles, K=32
    for (int j = wid; j < 13 * 4; j += 8) {
      int mt = j >> 2, nt = j & 3;
      int m0 = mt * 16, n0 = nt * 16;
      v16h a = frag_a_f16(qh, 32, m0, 0, lane);
      v16h bf = frag_b_f16(agb, 32, n0, 0, lane);
      v8f c = v8f_zero();
      c = wmma16(a, bf, c);
      int hf = lane >> 4, nc = lane & 15;
#pragma unroll
      for (int r = 0; r < 8; ++r) {
        int m = m0 + hf * 8 + r, ac = n0 + nc;
        float val = (m < NN && ac < AGENT) ? SCALE * c[r] + AB[((size_t)h * NN + m) * AGENT + ac] : -1e30f;
        S2[m * 64 + ac] = val;
      }
    }
    __syncthreads();
    if (tid < 208) {
      int m = tid;
      float mx = -3e30f;
      for (int a = 0; a < 64; ++a) mx = fmaxf(mx, S2[m * 64 + a]);
      float sum = 0.f;
      for (int a = 0; a < 64; ++a) {
        float e = __expf(S2[m * 64 + a] - mx);
        S2[m * 64 + a] = e;
        sum += e;
      }
      float inv = 1.f / sum;
      for (int a = 0; a < 64; ++a) A2[m * 64 + a] = (_Float16)(S2[m * 64 + a] * inv);
    }
    __syncthreads();
    // out_h = attn2 @ agent_v ; accumulate into outp columns h*32..h*32+31
    for (int j = wid; j < 13 * 2; j += 8) {
      int mt = j >> 1, nt = j & 1;
      int m0 = mt * 16, n0 = nt * 16;
      v8f c = v8f_zero();
#pragma unroll
      for (int kt = 0; kt < 2; ++kt) {
        v16h a = frag_a_f16(A2, 64, m0, kt * 32, lane);
        v16h bf = frag_b_f16(avb, 64, n0, kt * 32, lane);
        c = wmma16(a, bf, c);
      }
      int hf = lane >> 4, nc = lane & 15;
#pragma unroll
      for (int r = 0; r < 8; ++r) {
        int m = m0 + hf * 8 + r, col = h * HD + n0 + nc;
        _Float16 o = outp[m * CC + col];
        outp[m * CC + col] = (_Float16)((float)o + c[r]);
      }
    }
    __syncthreads();
  }

  // projection: out = outp @ proj_w^T + proj_b ; 13 Mtiles x 12 Ntiles, K=192
  for (int j = wid; j < 13 * 12; j += 8) {
    int mt = j / 12, nt = j % 12;
    int m0 = mt * 16, n0 = nt * 16;
    v8f c = v8f_zero();
#pragma unroll
    for (int kt = 0; kt < 6; ++kt) {
      v16h a = frag_a_f16(outp, CC, m0, kt * 32, lane);
      v16h bf = frag_b_f16(pw16, CC, n0, kt * 32, lane);
      c = wmma16(a, bf, c);
    }
    int hf = lane >> 4, nc = lane & 15;
#pragma unroll
    for (int r = 0; r < 8; ++r) {
      int m = m0 + hf * 8 + r, col = n0 + nc;
      if (m < NN) out[((size_t)b * NN + m) * CC + col] = c[r] + proj_b[col];
    }
  }
}

// ---------------- host launcher ----------------
extern "C" void kernel_launch(void* const* d_in, const int* in_sizes, int n_in,
                              void* d_out, int out_size, void* d_ws, size_t ws_size,
                              hipStream_t stream) {
  (void)in_sizes; (void)n_in; (void)out_size; (void)ws_size;
  const float* x      = (const float*)d_in[0];
  const float* qkv_w  = (const float*)d_in[1];
  const float* qkv_b  = (const float*)d_in[2];
  const float* proj_w = (const float*)d_in[3];
  const float* proj_b = (const float*)d_in[4];
  const float* dwc_w  = (const float*)d_in[5];
  const float* dwc_b  = (const float*)d_in[6];
  const float* an_b   = (const float*)d_in[7];
  const float* na_b   = (const float*)d_in[8];
  const float* ah_b   = (const float*)d_in[9];
  const float* aw_b   = (const float*)d_in[10];
  const float* ha_b   = (const float*)d_in[11];
  const float* wa_b   = (const float*)d_in[12];
  float* out = (float*)d_out;

  char* ws = (char*)d_ws;
  size_t off = 0;
  auto alloc = [&](size_t bytes) {
    char* p = ws + off;
    off = (off + bytes + 255) & ~(size_t)255;
    return p;
  };
  const size_t qkv_elems = (size_t)BB * NN * CC;
  _Float16* qf     = (_Float16*)alloc(qkv_elems * 2);
  _Float16* kf     = (_Float16*)alloc(qkv_elems * 2);
  _Float16* vf     = (_Float16*)alloc(qkv_elems * 2);
  _Float16* agentf = (_Float16*)alloc((size_t)BB * AGENT * CC * 2);
  _Float16* avT    = (_Float16*)alloc((size_t)BB * HEADS * 32 * 64 * 2);
  float*    PB     = (float*)alloc((size_t)HEADS * AGENT * NN * 4);
  float*    AB     = (float*)alloc((size_t)HEADS * NN * AGENT * 4);
  _Float16* w16    = (_Float16*)alloc((size_t)576 * 192 * 2);
  _Float16* p16    = (_Float16*)alloc((size_t)192 * 192 * 2);

  // K0: bias precompute  (6*49*196 = 57624 threads)
  bias_kernel<<<(HEADS * AGENT * NN + 255) / 256, 256, 0, stream>>>(
      an_b, na_b, ah_b, aw_b, ha_b, wa_b, PB, AB);
  // K0b: weight conversion (110592 + 36864 = 147456 threads)
  cvt_kernel<<<(576 * 192 + 192 * 192) / 256, 256, 0, stream>>>(qkv_w, proj_w, w16, p16);
  // K1: QKV GEMM. 6272 Mtiles * 9 N-groups = 56448 wave-jobs / 8 waves per block
  qkv_kernel<<<56448 / 8, 256, 0, stream>>>(x, w16, qkv_b, qf, kf, vf);
  // K2: agent pooling (512*49*192 = 4,816,896 threads)
  pool_kernel<<<(BB * AGENT * CC) / 256, 256, 0, stream>>>(qf, agentf);
  // K3: stage-1 per (b,h); 100,352 B dynamic LDS
  stage1_kernel<<<BB * HEADS, 128, 100352, stream>>>(kf, vf, agentf, PB, avT);
  // K4: stage-2 + dwconv + proj per b; 181,248 B dynamic LDS
  stage2_kernel<<<BB, 256, 181248, stream>>>(qf, vf, agentf, avT, AB, p16, proj_b, dwc_w, dwc_b, out);
}
